// SmashRNNModel_44066364457499
// MI455X (gfx1250) — compile-verified, matrix operand order checked by reading, not run on hardware
//
#include <hip/hip_runtime.h>
#include <math.h>

// ---------------------------------------------------------------------------
// SmashRNN forward on gfx1250: bf16 WMMA (v_wmma_f32_16x16x32_bf16) for all
// GEMMs, one wave32 per 16-sequence tile, fused online-softmax attention.
// Round 2: A-fragments pre-packed to LDS once per GEMM (reused across all
// n-tiles); inner loop = 1 global B load + 1 WMMA. Biases staged in LDS.
// ---------------------------------------------------------------------------

typedef __attribute__((ext_vector_type(16))) __bf16 v16bf;
typedef __attribute__((ext_vector_type(8)))  float  v8f;
typedef __attribute__((ext_vector_type(8)))  unsigned int v8u;

__device__ __forceinline__ unsigned short f2bf_us(float f) {
  unsigned u = __builtin_bit_cast(unsigned, f);
  unsigned r = u + 0x7fffu + ((u >> 16) & 1u);   // round-to-nearest-even
  return (unsigned short)(r >> 16);
}
__device__ __forceinline__ float bf2f(unsigned short s) {
  unsigned u = ((unsigned)s) << 16;
  return __builtin_bit_cast(float, u);
}
__device__ __forceinline__ float sigmoidf_(float x) { return 1.f / (1.f + expf(-x)); }

// ---------------------------------------------------------------------------
// Pack a [16,K] f32 LDS operand into WMMA A-fragment layout (bf16), one v8u
// (32 bytes = 16 bf16) per lane per k-tile. Lane layout for 16-bit A 16x32:
// m = lane&15, half = lane>>4, element e holds k = e + (e&8) + 8*half.
// ---------------------------------------------------------------------------
__device__ __forceinline__ void pack_a(const float* __restrict__ A, int lda, int KT,
                                       v8u* __restrict__ apk, int lane) {
  const int mlo  = lane & 15;
  const int half = lane >> 4;
  for (int kt = 0; kt < KT; ++kt) {
    v8u w;
#pragma unroll
    for (int i = 0; i < 8; ++i) {
      int e0 = 2 * i;
      int k0 = kt * 32 + e0 + (e0 & 8) + 8 * half;   // k1 = k0 + 1 (contiguous pair)
      unsigned lo = f2bf_us(A[mlo * lda + k0]);
      unsigned hi = f2bf_us(A[mlo * lda + k0 + 1]);
      w[i] = lo | (hi << 16);
    }
    apk[kt * 32 + lane] = w;
  }
}

// ---------------------------------------------------------------------------
// [16,K] (pre-packed bf16 A-fragments in LDS) x packed-B (bf16, global/L2)
// -> [16,N] f32 in LDS via WMMA. KT compile-time so A fragments live in regs.
// ---------------------------------------------------------------------------
template<int KT>
__device__ __forceinline__ void gemm16(const v8u* __restrict__ apk,
                                       const unsigned short* __restrict__ Bp, int NTloc,
                                       float* __restrict__ C, int ldc, int lane) {
  const int mlo  = lane & 15;
  const int half = lane >> 4;
  v16bf ar[KT];
#pragma unroll
  for (int kt = 0; kt < KT; ++kt)
    ar[kt] = __builtin_bit_cast(v16bf, apk[kt * 32 + lane]);
  for (int nt = 0; nt < NTloc; ++nt) {
    v8f acc = {};
#pragma unroll
    for (int kt = 0; kt < KT; ++kt) {
      const v8u* bp = (const v8u*)(Bp + ((size_t)(kt * NTloc + nt) * 32 + (size_t)lane) * 16);
      v16bf bv = __builtin_bit_cast(v16bf, *bp);
      acc = __builtin_amdgcn_wmma_f32_16x16x32_bf16(false, ar[kt], false, bv,
                                                    (short)0, acc, false, false);
    }
#pragma unroll
    for (int r = 0; r < 8; ++r)
      C[(r + 8 * half) * ldc + nt * 16 + mlo] = acc[r];
  }
}

// ---------------------------------------------------------------------------
// Pack W [Nact,Kact] (row-major f32, used as B = W^T, K x N) into WMMA
// B-fragment order: element index ((kt*NT + nt)*32 + lane)*16 + e holds
// bf16(W[n*Kact + k]) with n = nt*16 + (lane&15),
// k = kt*32 + e + (e&8) + 8*(lane>>4); zero-padded out of range.
// ---------------------------------------------------------------------------
__global__ void pack_b_kernel(const float* __restrict__ Wm, int Nact, int Kact,
                              int NTloc, int KT, unsigned short* __restrict__ outp) {
  long idx = (long)blockIdx.x * blockDim.x + threadIdx.x;
  long total = (long)KT * NTloc * 512;
  if (idx >= total) return;
  int e    = (int)(idx & 15);
  int lane = (int)((idx >> 4) & 31);
  long tile = idx >> 9;
  int nt = (int)(tile % NTloc);
  int kt = (int)(tile / NTloc);
  int n = nt * 16 + (lane & 15);
  int k = kt * 32 + e + (e & 8) + 8 * (lane >> 4);
  float v = (n < Nact && k < Kact) ? Wm[(size_t)n * Kact + k] : 0.f;
  outp[idx] = f2bf_us(v);
}

// ---------------------------------------------------------------------------
// One bidirectional GRU + attention level. One wave32 per block handles 16
// sequences. Phase 1: forward GRU, store masked hidden states (bf16) to ws.
// Phase 2: backward GRU (per-row reversed valid prefix) + fused online-softmax
// attention pooling (order-invariant), writing pooled [N,2H].
// ---------------------------------------------------------------------------
struct LevelArgs {
  const float* emb;          // embedding table (word level) or nullptr
  const int*   ids;          // [N,T] ids (word level) or nullptr
  const float* xdir;         // [N,T,D] direct input (upper levels)
  const int*   lens;         // [N]
  const unsigned short *wih_f, *whh_f, *wih_b, *whh_b, *wa;  // packed bf16
  const float *bih_f, *bhh_f, *bih_b, *bhh_b, *ba, *v;
  unsigned short* hfstore;   // [N,T,H] bf16 forward hidden states
  float* pooled;             // [N,2H]
};

template<int X, int Y> struct CtMax { static constexpr int v = X > Y ? X : Y; };

template<int D, int H, int T>
__global__ __launch_bounds__(32)
void level_kernel(LevelArgs a) {
  constexpr int DPAD = (D + 31) / 32 * 32;
  constexpr int HPAD = (H + 31) / 32 * 32;
  constexpr int N3   = 3 * H;
  constexpr int NPAD = (N3 + 15) / 16 * 16;
  constexpr int NT   = NPAD / 16;
  constexpr int KTD  = DPAD / 32;
  constexpr int KTH  = HPAD / 32;
  constexpr int A2   = 2 * H;
  constexpr int A2P  = (A2 + 31) / 32 * 32;
  constexpr int KTA  = A2P / 32;
  constexpr int KTMAX = CtMax<CtMax<KTD, KTH>::v, KTA>::v;

  __shared__ float xrow[16 * DPAD];
  __shared__ float hbuf[16 * HPAD];
  __shared__ float gi[16 * NPAD];
  __shared__ float gh[16 * NPAD];
  __shared__ float outc[16 * A2P];
  __shared__ float proj[16 * 64];
  __shared__ float acc[16 * A2];
  __shared__ v8u   apk[KTMAX * 32];          // packed A fragments (bf16)
  __shared__ float bihs[NPAD], bhhs[NPAD];   // staged biases (per phase)
  __shared__ float bas[64], vs[64];          // attention bias / vector
  __shared__ float srow[16], Mrow[16], Zrow[16], facrow[16], wrow[16];
  __shared__ int   lenrow[16], tmrow[16];

  const int lane = threadIdx.x;
  const int ns0  = blockIdx.x * 16;

  if (lane < 16) lenrow[lane] = a.lens[ns0 + lane];
  for (int i = lane; i < 16 * HPAD; i += 32) hbuf[i] = 0.f;
  for (int j = lane; j < N3; j += 32) { bihs[j] = a.bih_f[j]; bhhs[j] = a.bhh_f[j]; }
  for (int j = lane; j < 64; j += 32) {
    bas[j] = (j < 50) ? a.ba[j] : 0.f;
    vs[j]  = (j < 50) ? a.v[j]  : 0.f;
  }
  __syncthreads();

  // ------------------ Phase 1: forward direction ------------------
  for (int t = 0; t < T; ++t) {
    for (int m = 0; m < 16; ++m) {
      const float* src;
      if (a.ids) src = a.emb + (size_t)a.ids[(size_t)(ns0 + m) * T + t] * D;
      else       src = a.xdir + ((size_t)(ns0 + m) * T + t) * (size_t)D;
      for (int k = lane; k < DPAD; k += 32)
        xrow[m * DPAD + k] = (k < D) ? src[k] : 0.f;
    }
    __syncthreads();
    pack_a(xrow, DPAD, KTD, apk, lane);
    __syncthreads();
    gemm16<KTD>(apk, a.wih_f, NT, gi, NPAD, lane);
    __syncthreads();
    pack_a(hbuf, HPAD, KTH, apk, lane);
    __syncthreads();
    gemm16<KTH>(apk, a.whh_f, NT, gh, NPAD, lane);
    __syncthreads();
    for (int m = 0; m < 16; ++m) {
      bool valid = t < lenrow[m];
      for (int j = lane; j < H; j += 32) {
        float ir  = gi[m * NPAD + j]         + bihs[j];
        float iz  = gi[m * NPAD + H + j]     + bihs[H + j];
        float inn = gi[m * NPAD + 2 * H + j] + bihs[2 * H + j];
        float hr  = gh[m * NPAD + j]         + bhhs[j];
        float hz  = gh[m * NPAD + H + j]     + bhhs[H + j];
        float hn  = gh[m * NPAD + 2 * H + j] + bhhs[2 * H + j];
        float r = sigmoidf_(ir + hr);
        float z = sigmoidf_(iz + hz);
        float n = tanhf(inn + r * hn);
        float hp = hbuf[m * HPAD + j];
        float hv = valid ? ((1.f - z) * n + z * hp) : hp;
        hbuf[m * HPAD + j] = hv;
        a.hfstore[((size_t)(ns0 + m) * T + t) * H + j] = f2bf_us(valid ? hv : 0.f);
      }
    }
    __syncthreads();
  }

  // --------- Phase 2: backward direction + online attention ---------
  for (int i = lane; i < 16 * HPAD; i += 32) hbuf[i] = 0.f;
  for (int i = lane; i < 16 * A2;   i += 32) acc[i]  = 0.f;
  for (int j = lane; j < N3; j += 32) { bihs[j] = a.bih_b[j]; bhhs[j] = a.bhh_b[j]; }
  if (lane < 16) { Mrow[lane] = -INFINITY; Zrow[lane] = 0.f; }
  __syncthreads();

  for (int i = 0; i < T; ++i) {
    if (lane < 16) {
      int len = lenrow[lane];
      tmrow[lane] = (i < len) ? (len - 1 - i) : i;   // reversed valid prefix
    }
    __syncthreads();
    for (int m = 0; m < 16; ++m) {
      int tt = tmrow[m];
      const float* src;
      if (a.ids) src = a.emb + (size_t)a.ids[(size_t)(ns0 + m) * T + tt] * D;
      else       src = a.xdir + ((size_t)(ns0 + m) * T + tt) * (size_t)D;
      for (int k = lane; k < DPAD; k += 32)
        xrow[m * DPAD + k] = (k < D) ? src[k] : 0.f;
    }
    __syncthreads();
    pack_a(xrow, DPAD, KTD, apk, lane);
    __syncthreads();
    gemm16<KTD>(apk, a.wih_b, NT, gi, NPAD, lane);
    __syncthreads();
    pack_a(hbuf, HPAD, KTH, apk, lane);
    __syncthreads();
    gemm16<KTH>(apk, a.whh_b, NT, gh, NPAD, lane);
    __syncthreads();
    for (int m = 0; m < 16; ++m) {
      bool valid = i < lenrow[m];
      for (int j = lane; j < H; j += 32) {
        float ir  = gi[m * NPAD + j]         + bihs[j];
        float iz  = gi[m * NPAD + H + j]     + bihs[H + j];
        float inn = gi[m * NPAD + 2 * H + j] + bihs[2 * H + j];
        float hr  = gh[m * NPAD + j]         + bhhs[j];
        float hz  = gh[m * NPAD + H + j]     + bhhs[H + j];
        float hn  = gh[m * NPAD + 2 * H + j] + bhhs[2 * H + j];
        float r = sigmoidf_(ir + hr);
        float z = sigmoidf_(iz + hz);
        float n = tanhf(inn + r * hn);
        float hp = hbuf[m * HPAD + j];
        float hv = valid ? ((1.f - z) * n + z * hp) : hp;
        hbuf[m * HPAD + j] = hv;
      }
    }
    __syncthreads();
    // out_t = [h_fwd(tm), h_bwd_i]
    for (int m = 0; m < 16; ++m) {
      const unsigned short* hf = a.hfstore + ((size_t)(ns0 + m) * T + tmrow[m]) * H;
      for (int k = lane; k < A2P; k += 32) {
        float vv = 0.f;
        if (k < H)        vv = bf2f(hf[k]);
        else if (k < A2)  vv = hbuf[m * HPAD + (k - H)];
        outc[m * A2P + k] = vv;
      }
    }
    __syncthreads();
    pack_a(outc, A2P, KTA, apk, lane);
    __syncthreads();
    gemm16<KTA>(apk, a.wa, 4, proj, 64, lane);   // [16,64] (ATT=50 valid)
    __syncthreads();
    if (lane < 16) {
      float s = 0.f;
      for (int j = 0; j < 50; ++j)
        s += vs[j] * tanhf(proj[lane * 64 + j] + bas[j]);
      srow[lane] = s;
      bool valid = i < lenrow[lane];
      float fac = 1.f, w = 0.f;
      if (valid) {
        float nm = fmaxf(Mrow[lane], s);
        fac = expf(Mrow[lane] - nm);
        w   = expf(s - nm);
        Zrow[lane] = Zrow[lane] * fac + w;
        Mrow[lane] = nm;
      }
      facrow[lane] = fac; wrow[lane] = w;
    }
    __syncthreads();
    for (int m = 0; m < 16; ++m) {
      float fac = facrow[m], w = wrow[m];
      if (fac == 1.f && w == 0.f) continue;   // invalid step: no contribution
      for (int k = lane; k < A2; k += 32)
        acc[m * A2 + k] = acc[m * A2 + k] * fac + w * outc[m * A2P + k];
    }
    __syncthreads();
  }

  for (int m = 0; m < 16; ++m) {
    float Z = Zrow[m];
    float inv = (Z > 0.f) ? 1.f / fmaxf(Z, 1e-30f) : 0.f;
    for (int k = lane; k < A2; k += 32)
      a.pooled[(size_t)(ns0 + m) * A2 + k] = acc[m * A2 + k] * inv;
  }
}

// ---------------------------------------------------------------------------
// Head: feat=[dc,dp,|dc-dp|] (1800), h1=relu(feat@W1^T+b1) (900),
// out=sigmoid(h1@W2^T+b2). One block per batch row.
// ---------------------------------------------------------------------------
__global__ void head_kernel(const float* __restrict__ para_pooled,
                            const float* __restrict__ W1, const float* __restrict__ b1,
                            const float* __restrict__ W2, const float* __restrict__ b2,
                            float* __restrict__ out) {
  __shared__ float feat[1800];
  __shared__ float h1[900];
  __shared__ float red[256];
  int b = blockIdx.x;
  const float* dc = para_pooled + (size_t)b * 600;
  const float* dp = para_pooled + (size_t)(8 + b) * 600;
  for (int k = threadIdx.x; k < 600; k += blockDim.x) {
    float c = dc[k], p = dp[k];
    feat[k] = c; feat[600 + k] = p; feat[1200 + k] = fabsf(c - p);
  }
  __syncthreads();
  for (int j = threadIdx.x; j < 900; j += blockDim.x) {
    float s = b1[j];
    const float* wr = W1 + (size_t)j * 1800;
    for (int k = 0; k < 1800; ++k) s += wr[k] * feat[k];
    h1[j] = fmaxf(s, 0.f);
  }
  __syncthreads();
  float partial = 0.f;
  for (int j = threadIdx.x; j < 900; j += blockDim.x) partial += W2[j] * h1[j];
  red[threadIdx.x] = partial;
  __syncthreads();
  for (int s = 128; s > 0; s >>= 1) {
    if ((int)threadIdx.x < s) red[threadIdx.x] += red[threadIdx.x + s];
    __syncthreads();
  }
  if (threadIdx.x == 0) out[b] = 1.f / (1.f + expf(-(red[0] + b2[0])));
}

// ---------------------------------------------------------------------------
extern "C" void kernel_launch(void* const* d_in, const int* in_sizes, int n_in,
                              void* d_out, int out_size, void* d_ws, size_t ws_size,
                              hipStream_t stream) {
  (void)in_sizes; (void)n_in; (void)out_size; (void)ws_size;
  const int B = 8, P = 10, S = 16, Wt = 48;
  const int NW = 2 * B * P * S;   // 2560 word sequences (cur+prev)
  const int NS = 2 * B * P;       // 160 sentence sequences
  const int NP = 2 * B;           // 16 paragraph sequences

  const int* doc_c = (const int*)d_in[0];
  const int* wps_c = (const int*)d_in[1];
  const int* spp_c = (const int*)d_in[2];
  const int* ppd_c = (const int*)d_in[3];
  const int* doc_p = (const int*)d_in[4];
  const int* wps_p = (const int*)d_in[5];
  const int* spp_p = (const int*)d_in[6];
  const int* ppd_p = (const int*)d_in[7];
  // d_in[8] = click_rate_tensor (unused by the reference forward)
  const float* emb = (const float*)d_in[9];

  struct GruIn { const float *Wih_f,*Whh_f,*bih_f,*bhh_f,*Wih_b,*Whh_b,*bih_b,*bhh_b,*Wa,*ba,*v; };
  int pi = 10;
  auto grab = [&](GruIn& g) {
    g.Wih_f=(const float*)d_in[pi++]; g.Whh_f=(const float*)d_in[pi++];
    g.bih_f=(const float*)d_in[pi++]; g.bhh_f=(const float*)d_in[pi++];
    g.Wih_b=(const float*)d_in[pi++]; g.Whh_b=(const float*)d_in[pi++];
    g.bih_b=(const float*)d_in[pi++]; g.bhh_b=(const float*)d_in[pi++];
    g.Wa=(const float*)d_in[pi++];    g.ba=(const float*)d_in[pi++];
    g.v=(const float*)d_in[pi++];
  };
  GruIn gw, gs, gp; grab(gw); grab(gs); grab(gp);
  const float* W1 = (const float*)d_in[pi++];
  const float* b1 = (const float*)d_in[pi++];
  const float* W2 = (const float*)d_in[pi++];
  const float* b2 = (const float*)d_in[pi++];

  char* base = (char*)d_ws;
  size_t off = 0;
  auto alloc = [&](size_t bytes) -> void* {
    off = (off + 255) & ~(size_t)255;
    void* p = base + off;
    off += bytes;
    return p;
  };
  auto rup = [](int x, int m) { return (x + m - 1) / m * m; };

  // Level tiling constants (must match level_kernel template math)
  struct LC { int D, H, KTD, KTH, NT, KTA; };
  LC Lw{200, 100, 0, 0, 0, 0}, Ls{200, 200, 0, 0, 0, 0}, Lp{400, 300, 0, 0, 0, 0};
  auto fill = [&](LC& c) {
    c.KTD = rup(c.D, 32) / 32;
    c.KTH = rup(c.H, 32) / 32;
    c.NT  = rup(3 * c.H, 16) / 16;
    c.KTA = rup(2 * c.H, 32) / 32;
  };
  fill(Lw); fill(Ls); fill(Lp);

  int* ids_w  = (int*)alloc((size_t)NW * Wt * 4);
  int* lens_w = (int*)alloc((size_t)NW * 4);
  int* lens_s = (int*)alloc((size_t)NS * 4);
  int* lens_p = (int*)alloc((size_t)NP * 4);

  auto allocPacked = [&](int KT, int NT) {
    return (unsigned short*)alloc((size_t)KT * NT * 512 * 2);
  };
  unsigned short *wW_ihf = allocPacked(Lw.KTD, Lw.NT), *wW_hhf = allocPacked(Lw.KTH, Lw.NT),
                 *wW_ihb = allocPacked(Lw.KTD, Lw.NT), *wW_hhb = allocPacked(Lw.KTH, Lw.NT),
                 *wW_a   = allocPacked(Lw.KTA, 4);
  unsigned short *sW_ihf = allocPacked(Ls.KTD, Ls.NT), *sW_hhf = allocPacked(Ls.KTH, Ls.NT),
                 *sW_ihb = allocPacked(Ls.KTD, Ls.NT), *sW_hhb = allocPacked(Ls.KTH, Ls.NT),
                 *sW_a   = allocPacked(Ls.KTA, 4);
  unsigned short *pW_ihf = allocPacked(Lp.KTD, Lp.NT), *pW_hhf = allocPacked(Lp.KTH, Lp.NT),
                 *pW_ihb = allocPacked(Lp.KTD, Lp.NT), *pW_hhb = allocPacked(Lp.KTH, Lp.NT),
                 *pW_a   = allocPacked(Lp.KTA, 4);

  unsigned short* hf_w = (unsigned short*)alloc((size_t)NW * 48 * 100 * 2);
  unsigned short* hf_s = (unsigned short*)alloc((size_t)NS * 16 * 200 * 2);
  unsigned short* hf_p = (unsigned short*)alloc((size_t)NP * 10 * 300 * 2);
  float* pool_w = (float*)alloc((size_t)NW * 200 * 4);
  float* pool_s = (float*)alloc((size_t)NS * 400 * 4);
  float* pool_p = (float*)alloc((size_t)NP * 600 * 4);

  // Concatenate (cur, prev) ids and lengths (device-to-device, graph-safe).
  size_t docN = (size_t)B * P * S * Wt;
  hipMemcpyAsync(ids_w,        doc_c, docN * 4, hipMemcpyDeviceToDevice, stream);
  hipMemcpyAsync(ids_w + docN, doc_p, docN * 4, hipMemcpyDeviceToDevice, stream);
  hipMemcpyAsync(lens_w,             wps_c, (size_t)B * P * S * 4, hipMemcpyDeviceToDevice, stream);
  hipMemcpyAsync(lens_w + B * P * S, wps_p, (size_t)B * P * S * 4, hipMemcpyDeviceToDevice, stream);
  hipMemcpyAsync(lens_s,         spp_c, (size_t)B * P * 4, hipMemcpyDeviceToDevice, stream);
  hipMemcpyAsync(lens_s + B * P, spp_p, (size_t)B * P * 4, hipMemcpyDeviceToDevice, stream);
  hipMemcpyAsync(lens_p,     ppd_c, (size_t)B * 4, hipMemcpyDeviceToDevice, stream);
  hipMemcpyAsync(lens_p + B, ppd_p, (size_t)B * 4, hipMemcpyDeviceToDevice, stream);

  auto pack = [&](const float* Wm, int Nact, int Kact, int NT, int KT, unsigned short* outp) {
    long total = (long)KT * NT * 512;
    int blocks = (int)((total + 255) / 256);
    pack_b_kernel<<<blocks, 256, 0, stream>>>(Wm, Nact, Kact, NT, KT, outp);
  };
  // word: Wih [300,200], Whh [300,100], Wa [50,200]
  pack(gw.Wih_f, 300, 200, Lw.NT, Lw.KTD, wW_ihf);
  pack(gw.Whh_f, 300, 100, Lw.NT, Lw.KTH, wW_hhf);
  pack(gw.Wih_b, 300, 200, Lw.NT, Lw.KTD, wW_ihb);
  pack(gw.Whh_b, 300, 100, Lw.NT, Lw.KTH, wW_hhb);
  pack(gw.Wa,     50, 200, 4,     Lw.KTA, wW_a);
  // sent: Wih [600,200], Whh [600,200], Wa [50,400]
  pack(gs.Wih_f, 600, 200, Ls.NT, Ls.KTD, sW_ihf);
  pack(gs.Whh_f, 600, 200, Ls.NT, Ls.KTH, sW_hhf);
  pack(gs.Wih_b, 600, 200, Ls.NT, Ls.KTD, sW_ihb);
  pack(gs.Whh_b, 600, 200, Ls.NT, Ls.KTH, sW_hhb);
  pack(gs.Wa,     50, 400, 4,     Ls.KTA, sW_a);
  // para: Wih [900,400], Whh [900,300], Wa [50,600]
  pack(gp.Wih_f, 900, 400, Lp.NT, Lp.KTD, pW_ihf);
  pack(gp.Whh_f, 900, 300, Lp.NT, Lp.KTH, pW_hhf);
  pack(gp.Wih_b, 900, 400, Lp.NT, Lp.KTD, pW_ihb);
  pack(gp.Whh_b, 900, 300, Lp.NT, Lp.KTH, pW_hhb);
  pack(gp.Wa,     50, 600, 4,     Lp.KTA, pW_a);

  // Word level: embedding-gather input
  LevelArgs aw{};
  aw.emb = emb; aw.ids = ids_w; aw.xdir = nullptr; aw.lens = lens_w;
  aw.wih_f = wW_ihf; aw.whh_f = wW_hhf; aw.wih_b = wW_ihb; aw.whh_b = wW_hhb; aw.wa = wW_a;
  aw.bih_f = gw.bih_f; aw.bhh_f = gw.bhh_f; aw.bih_b = gw.bih_b; aw.bhh_b = gw.bhh_b;
  aw.ba = gw.ba; aw.v = gw.v;
  aw.hfstore = hf_w; aw.pooled = pool_w;
  level_kernel<200, 100, 48><<<NW / 16, 32, 0, stream>>>(aw);

  // Sentence level: input = word pooled [2560,200] viewed [160,16,200]
  LevelArgs as{};
  as.emb = nullptr; as.ids = nullptr; as.xdir = pool_w; as.lens = lens_s;
  as.wih_f = sW_ihf; as.whh_f = sW_hhf; as.wih_b = sW_ihb; as.whh_b = sW_hhb; as.wa = sW_a;
  as.bih_f = gs.bih_f; as.bhh_f = gs.bhh_f; as.bih_b = gs.bih_b; as.bhh_b = gs.bhh_b;
  as.ba = gs.ba; as.v = gs.v;
  as.hfstore = hf_s; as.pooled = pool_s;
  level_kernel<200, 200, 16><<<NS / 16, 32, 0, stream>>>(as);

  // Paragraph level: input = sentence pooled [160,400] viewed [16,10,400]
  LevelArgs ap{};
  ap.emb = nullptr; ap.ids = nullptr; ap.xdir = pool_s; ap.lens = lens_p;
  ap.wih_f = pW_ihf; ap.whh_f = pW_hhf; ap.wih_b = pW_ihb; ap.whh_b = pW_hhb; ap.wa = pW_a;
  ap.bih_f = gp.bih_f; ap.bhh_f = gp.bhh_f; ap.bih_b = gp.bih_b; ap.bhh_b = gp.bhh_b;
  ap.ba = gp.ba; ap.v = gp.v;
  ap.hfstore = hf_p; ap.pooled = pool_p;
  level_kernel<400, 300, 10><<<NP / 16, 32, 0, stream>>>(ap);

  // Head MLP -> sigmoid, rows 0-7 = current doc, 8-15 = previous doc
  head_kernel<<<8, 256, 0, stream>>>(pool_p, W1, b1, W2, b2, (float*)d_out);
}
